// ReducedGlobalAttention_81200651698440
// MI455X (gfx1250) — matrix operand order repaired and checked by reading, not compiled, not run on hardware
//
#include <hip/hip_runtime.h>
#include <hip/hip_bf16.h>
#include <math.h>

// ---------------------------------------------------------------------------
// CDNA5 (gfx1250) wave32 WMMA types & helpers
// ---------------------------------------------------------------------------
typedef __bf16 v16bf __attribute__((ext_vector_type(16)));
typedef float  v8f   __attribute__((ext_vector_type(8)));

__device__ __forceinline__ v8f wmma_bf16(v16bf a, v16bf b, v8f c) {
    // D = A(16x32 bf16) * B(32x16 bf16) + C(16x16 f32)
    return __builtin_amdgcn_wmma_f32_16x16x32_bf16(false, a, false, b,
                                                   (short)0, c, false, false);
}

// A-fragment (16x32 bf16). p points at K = (lane>=16 ? 16 : 0) within the 32-chunk.
// Layout: lanes 0-15 hold M=lane, K 0-15; lanes 16-31 hold K 16-31.
__device__ __forceinline__ v16bf load_a16(const __bf16* p) {
    v16bf a;
#pragma unroll
    for (int e = 0; e < 16; ++e) a[e] = p[e];
    return a;
}

// B-fragment (32x16 bf16). p points at K = (lane>=16 ? 16 : 0); lane&15 = N.
__device__ __forceinline__ v16bf load_b16(const __bf16* p) {
    v16bf b;
#pragma unroll
    for (int e = 0; e < 16; ++e) b[e] = p[e];
    return b;
}

// 16-lane (half-wave) reductions: xor masks 1..8 stay inside each 16-lane group.
__device__ __forceinline__ float half16_max(float v) {
#pragma unroll
    for (int m = 1; m < 16; m <<= 1) v = fmaxf(v, __shfl_xor(v, m, 32));
    return v;
}
__device__ __forceinline__ float half16_sum(float v) {
#pragma unroll
    for (int m = 1; m < 16; m <<= 1) v += __shfl_xor(v, m, 32);
    return v;
}

// Problem constants
#define NB   2
#define CIN  512
#define KC   256
#define VC   256
#define COUT 512
#define HW   4096
#define WDIM 64
#define POOLED 110   // 1 + 9 + 36 + 64 bins per channel

// ---------------------------------------------------------------------------
// Prep: weights -> bf16 (fused qkv weight), BN scale/shift folding
// ---------------------------------------------------------------------------
__global__ void prep_kernel(const float* wq, const float* wk, const float* wv,
                            const float* wpspk, const float* wpspv, const float* wW,
                            const float* gq, const float* bq, const float* mq, const float* vq,
                            const float* gk, const float* bk, const float* mk, const float* vk,
                            __bf16* W3bf, __bf16* wpspk_bf, __bf16* wpspv_bf, __bf16* wWbf,
                            float* bnq_s, float* bnq_b, float* bnk_s, float* bnk_b) {
    const int total = 768 * CIN;  // largest array
    for (int i = blockIdx.x * blockDim.x + threadIdx.x; i < total;
         i += gridDim.x * blockDim.x) {
        int row = i >> 9;  // /512
        float w = (row < 256) ? wq[i] : (row < 512 ? wk[i - 256 * CIN]
                                                   : wv[i - 512 * CIN]);
        W3bf[i] = (__bf16)w;
        if (i < 256 * 1024) { wpspk_bf[i] = (__bf16)wpspk[i]; wpspv_bf[i] = (__bf16)wpspv[i]; }
        if (i < COUT * VC)  { wWbf[i] = (__bf16)wW[i]; }
        if (i < 256) {
            float sq = gq[i] * rsqrtf(vq[i] + 1e-5f);
            bnq_s[i] = sq; bnq_b[i] = bq[i] - mq[i] * sq;
            float sk = gk[i] * rsqrtf(vk[i] + 1e-5f);
            bnk_s[i] = sk; bnk_b[i] = bk[i] - mk[i] * sk;
        }
    }
}

// ---------------------------------------------------------------------------
// GEMM 1: fused q/k/v 1x1 conv.  Y[768 x 4096] = W3[768 x 512] * x[n][512 x 4096]
// 64x64 register blocking per wave: 4 A-frags x 4 B-frags -> 16 WMMA / K-chunk.
// Epilogue: rows 0-255  -> BN+ReLU -> qT (p-major bf16)
//           rows 256-511-> BN+ReLU -> kpre (c-major f32, feeds pooling)
//           rows 512-767-> vpre (c-major f32, feeds pooling)
// ---------------------------------------------------------------------------
__global__ void __launch_bounds__(256)
gemm_qkv(const float* __restrict__ x, const __bf16* __restrict__ W3bf,
         const float* bnq_s, const float* bnq_b,
         const float* bnk_s, const float* bnk_b,
         __bf16* __restrict__ qT, float* __restrict__ kpre, float* __restrict__ vpre) {
    const int wave = threadIdx.x >> 5, lane = threadIdx.x & 31;
    const int unit = blockIdx.x * 8 + wave;          // 2 * 12 * 64 units
    const int n    = unit / (12 * 64);
    const int rem  = unit % (12 * 64);
    const int m0   = (rem % 12) * 64;
    const int p0   = (rem / 12) * 64;
    const int l15  = lane & 15;
    const int hi   = (lane >= 16);
    const float* xb = x + (size_t)n * CIN * HW;

    v8f acc[4][4] = {};
    for (int k0 = 0; k0 < CIN; k0 += 32) {
        v16bf a[4], b[4];
#pragma unroll
        for (int mi = 0; mi < 4; ++mi)
            a[mi] = load_a16(W3bf + (size_t)(m0 + mi * 16 + l15) * CIN + k0 + (hi ? 16 : 0));
#pragma unroll
        for (int pi = 0; pi < 4; ++pi) {
            const float* bcol = xb + (size_t)(k0 + (hi ? 16 : 0)) * HW + p0 + pi * 16 + l15;
#pragma unroll
            for (int e = 0; e < 16; ++e) b[pi][e] = (__bf16)bcol[(size_t)e * HW];
        }
#pragma unroll
        for (int mi = 0; mi < 4; ++mi)
#pragma unroll
            for (int pi = 0; pi < 4; ++pi)
                acc[mi][pi] = wmma_bf16(a[mi], b[pi], acc[mi][pi]);
    }
#pragma unroll
    for (int mi = 0; mi < 4; ++mi)
#pragma unroll
        for (int pi = 0; pi < 4; ++pi)
#pragma unroll
            for (int r = 0; r < 8; ++r) {
                int m = m0 + mi * 16 + r + (hi ? 8 : 0);
                int p = p0 + pi * 16 + l15;
                float v = acc[mi][pi][r];
                if (m < 256) {
                    v = fmaxf(v * bnq_s[m] + bnq_b[m], 0.0f);
                    qT[((size_t)n * HW + p) * KC + m] = (__bf16)v;
                } else if (m < 512) {
                    int c = m - 256;
                    v = fmaxf(v * bnk_s[c] + bnk_b[c], 0.0f);
                    kpre[((size_t)n * KC + c) * HW + p] = v;
                } else {
                    int c = m - 512;
                    vpre[((size_t)n * VC + c) * HW + p] = v;
                }
            }
}

// ---------------------------------------------------------------------------
// Adaptive average pool (PyTorch bin boundaries) for kpre & vpre.
// pooled[n][c][bin], bins packed s=1(1) | s=3(9) | s=6(36) | s=8(64) = 110.
// ---------------------------------------------------------------------------
__global__ void pool_kernel(const float* __restrict__ kpre, const float* __restrict__ vpre,
                            float* __restrict__ pooled_k, float* __restrict__ pooled_v) {
    const int TOT = 2 * NB * 256 * POOLED;
    int idx = blockIdx.x * blockDim.x + threadIdx.x;
    if (idx >= TOT) return;
    int t   = idx / (NB * 256 * POOLED);
    int r   = idx % (NB * 256 * POOLED);
    int n   = r / (256 * POOLED);
    int r2  = r % (256 * POOLED);
    int c   = r2 / POOLED;
    int bin = r2 % POOLED;
    int s, off;
    if (bin < 1)       { s = 1; off = 0; }
    else if (bin < 10) { s = 3; off = 1; }
    else if (bin < 46) { s = 6; off = 10; }
    else               { s = 8; off = 46; }
    int local = bin - off;
    int by = local / s, bx = local % s;
    int ly = (by * WDIM) / s, hy = ((by + 1) * WDIM + s - 1) / s;
    int lx = (bx * WDIM) / s, hx = ((bx + 1) * WDIM + s - 1) / s;
    const float* src = (t ? vpre : kpre) + ((size_t)n * 256 + c) * HW;
    float sum = 0.0f;
    for (int y = ly; y < hy; ++y)
        for (int xx = lx; xx < hx; ++xx) sum += src[y * WDIM + xx];
    float mean = sum / (float)((hy - ly) * (hx - lx));
    (t ? pooled_v : pooled_k)[r] = mean;
}

// ---------------------------------------------------------------------------
// Bilinear upsample (half-pixel, align_corners=False) + channel concat.
// Output catT[n][p][1024] bf16 (pixel-major so PSP GEMM B-loads are contiguous).
// ---------------------------------------------------------------------------
__global__ void upcat_kernel(const float* __restrict__ pooled_k,
                             const float* __restrict__ pooled_v,
                             __bf16* __restrict__ catTk, __bf16* __restrict__ catTv) {
    const size_t TOT = (size_t)2 * NB * HW * 1024;
    for (size_t idx = (size_t)blockIdx.x * blockDim.x + threadIdx.x; idx < TOT;
         idx += (size_t)gridDim.x * blockDim.x) {
        int t  = (int)(idx / ((size_t)NB * HW * 1024));
        size_t r = idx % ((size_t)NB * HW * 1024);
        int n  = (int)(r / (HW * 1024));
        int r2 = (int)(r % (HW * 1024));
        int p  = r2 / 1024;
        int ch = r2 % 1024;
        int si = ch >> 8, c = ch & 255;
        const int SS[4]  = {1, 3, 6, 8};
        const int OFF[4] = {0, 1, 10, 46};
        int s = SS[si], off = OFF[si];
        int y = p >> 6, xx = p & 63;
        const float* pool = (t ? pooled_v : pooled_k) + ((size_t)n * 256 + c) * POOLED + off;
        float val;
        if (s == 1) {
            val = pool[0];
        } else {
            float sy = (y + 0.5f) * s * (1.0f / WDIM) - 0.5f;
            float sx = (xx + 0.5f) * s * (1.0f / WDIM) - 0.5f;
            int y0 = (int)floorf(sy); float fy = sy - (float)y0;
            int x0 = (int)floorf(sx); float fx = sx - (float)x0;
            int y0c = min(max(y0, 0), s - 1), y1c = min(max(y0 + 1, 0), s - 1);
            int x0c = min(max(x0, 0), s - 1), x1c = min(max(x0 + 1, 0), s - 1);
            float v00 = pool[y0c * s + x0c], v01 = pool[y0c * s + x1c];
            float v10 = pool[y1c * s + x0c], v11 = pool[y1c * s + x1c];
            val = (1.0f - fy) * ((1.0f - fx) * v00 + fx * v01) +
                  fy         * ((1.0f - fx) * v10 + fx * v11);
        }
        (t ? catTv : catTk)[((size_t)n * HW + p) * 1024 + ch] = (__bf16)val;
    }
}

// ---------------------------------------------------------------------------
// GEMM 2 (PSP): Y[256 x 4096] = Wpsp[256 x 1024] * catT + bias.
// 64x64 register blocking. t=0 -> kT (pixel-major); t=1 -> v (channel-major).
// ---------------------------------------------------------------------------
__global__ void __launch_bounds__(256)
gemm_psp(const __bf16* __restrict__ wk_bf, const __bf16* __restrict__ wv_bf,
         const float* __restrict__ bk, const float* __restrict__ bv,
         const __bf16* __restrict__ catTk, const __bf16* __restrict__ catTv,
         __bf16* __restrict__ kT, __bf16* __restrict__ vmat) {
    const int wave = threadIdx.x >> 5, lane = threadIdx.x & 31;
    const int unit = blockIdx.x * 8 + wave;          // 2 tensors * 2 batches * 4 * 64
    const int t    = unit / (4 * 64 * NB);
    const int rem  = unit % (4 * 64 * NB);
    const int n    = rem / (4 * 64);
    const int rem2 = rem % (4 * 64);
    const int m0   = (rem2 % 4) * 64;
    const int p0   = (rem2 / 4) * 64;
    const int l15  = lane & 15, hi = (lane >= 16);
    const __bf16* W    = t ? wv_bf : wk_bf;
    const __bf16* catT = (t ? catTv : catTk) + (size_t)n * HW * 1024;

    v8f acc[4][4] = {};
    for (int k0 = 0; k0 < 1024; k0 += 32) {
        v16bf a[4], b[4];
#pragma unroll
        for (int mi = 0; mi < 4; ++mi)
            a[mi] = load_a16(W + (size_t)(m0 + mi * 16 + l15) * 1024 + k0 + (hi ? 16 : 0));
#pragma unroll
        for (int pi = 0; pi < 4; ++pi)
            b[pi] = load_b16(catT + (size_t)(p0 + pi * 16 + l15) * 1024 + k0 + (hi ? 16 : 0));
#pragma unroll
        for (int mi = 0; mi < 4; ++mi)
#pragma unroll
            for (int pi = 0; pi < 4; ++pi)
                acc[mi][pi] = wmma_bf16(a[mi], b[pi], acc[mi][pi]);
    }
    const float* bias = t ? bv : bk;
#pragma unroll
    for (int mi = 0; mi < 4; ++mi)
#pragma unroll
        for (int pi = 0; pi < 4; ++pi)
#pragma unroll
            for (int r = 0; r < 8; ++r) {
                int m = m0 + mi * 16 + r + (hi ? 8 : 0);
                int p = p0 + pi * 16 + l15;
                float v = acc[mi][pi][r] + bias[m];
                if (t) vmat[((size_t)n * VC + m) * HW + p] = (__bf16)v;
                else   kT[((size_t)n * HW + p) * KC + m]   = (__bf16)v;
            }
}

// ---------------------------------------------------------------------------
// Flash attention: block = 4 waves = 2 query tiles x 2 channel halves.
// Each wave: S for its 16-query tile (16 WMMA / 32-key chunk, shared work
// duplicated only 2x), online softmax, P via per-wave LDS (D->A layout with
// s_wait_dscnt), then 8 WMMA accumulating its 128 v-channels.
// ---------------------------------------------------------------------------
__global__ void __launch_bounds__(128)
attn_kernel(const __bf16* __restrict__ qT, const __bf16* __restrict__ kT,
            const __bf16* __restrict__ vmat, __bf16* __restrict__ ctxT) {
    __shared__ __bf16 ldsP[4][16 * 32];
    const int wave = threadIdx.x >> 5, lane = threadIdx.x & 31;
    const int qt   = blockIdx.x * 2 + (wave >> 1);   // 2*256 query tiles total
    const int q0   = (qt & 255) * 16;
    const int n    = qt >> 8;
    const int chalf = (wave & 1) * 128;
    const int l15  = lane & 15, hi = (lane >= 16);
    const __bf16* qb = qT   + (size_t)n * HW * KC;
    const __bf16* kb = kT   + (size_t)n * HW * KC;
    const __bf16* vb = vmat + (size_t)n * VC * HW;
    const float scale = 0.0625f;  // 1/sqrt(256)
    __bf16* pbuf = ldsP[wave];

    // Preload all q A-fragments (K = 256 channels = 8 chunks)
    v16bf aq[8];
#pragma unroll
    for (int kk = 0; kk < 8; ++kk)
        aq[kk] = load_a16(qb + (size_t)(q0 + l15) * KC + kk * 32 + (hi ? 16 : 0));

    v8f O[8] = {};
    float rowm[8], rows_[8];
#pragma unroll
    for (int r = 0; r < 8; ++r) { rowm[r] = -INFINITY; rows_[r] = 0.0f; }

    for (int p0 = 0; p0 < HW; p0 += 32) {
        // Prefetch next chunk of k rows (global_prefetch_b8; speculative, OOB-safe)
        __builtin_prefetch(kb + (size_t)(p0 + 32 + l15) * KC, 0, 0);
        __builtin_prefetch(kb + (size_t)(p0 + 48 + l15) * KC, 0, 0);

        // S tiles for key columns [p0, p0+16) and [p0+16, p0+32)
        v8f S0 = {}, S1 = {};
#pragma unroll
        for (int kk = 0; kk < 8; ++kk) {
            v16bf b0 = load_b16(kb + (size_t)(p0 + l15)      * KC + kk * 32 + (hi ? 16 : 0));
            v16bf b1 = load_b16(kb + (size_t)(p0 + 16 + l15) * KC + kk * 32 + (hi ? 16 : 0));
            S0 = wmma_bf16(aq[kk], b0, S0);
            S1 = wmma_bf16(aq[kk], b1, S1);
        }
        // Online softmax update + write P tile to LDS (D-layout positions)
#pragma unroll
        for (int r = 0; r < 8; ++r) {
            float s0 = S0[r] * scale, s1 = S1[r] * scale;
            float cm = half16_max(fmaxf(s0, s1));
            float nm = fmaxf(rowm[r], cm);
            float corr = __expf(rowm[r] - nm);
            rowm[r] = nm;
            float e0 = __expf(s0 - nm), e1 = __expf(s1 - nm);
            rows_[r] = rows_[r] * corr + half16_sum(e0 + e1);
#pragma unroll
            for (int t = 0; t < 8; ++t) O[t][r] *= corr;
            int m = r + (hi ? 8 : 0);
            pbuf[m * 32 + l15]      = (__bf16)e0;
            pbuf[m * 32 + 16 + l15] = (__bf16)e1;
        }
        // All lanes' LDS writes must land before the cross-lane A-layout reads.
        asm volatile("s_wait_dscnt 0x0" ::: "memory");
        v16bf pa = load_a16(pbuf + l15 * 32 + (hi ? 16 : 0));
        // O += P (16x32) * V^T (32 x 16 per channel tile); contiguous bf16 loads.
#pragma unroll
        for (int t = 0; t < 8; ++t) {
            v16bf bv_ = load_b16(vb + (size_t)(chalf + t * 16 + l15) * HW + p0 + (hi ? 16 : 0));
            O[t] = wmma_bf16(pa, bv_, O[t]);
        }
    }
    // Normalize and store ctxT[n][p][vc] (pixel-major for the final GEMM)
#pragma unroll
    for (int t = 0; t < 8; ++t)
#pragma unroll
        for (int r = 0; r < 8; ++r) {
            float val = O[t][r] / rows_[r];
            ctxT[((size_t)n * HW + q0 + r + (hi ? 8 : 0)) * VC + chalf + t * 16 + l15] =
                (__bf16)val;
        }
}

// ---------------------------------------------------------------------------
// GEMM 3: out[512 x 4096] = wW[512 x 256] * ctx  (f32 output), 64x64 blocking
// ---------------------------------------------------------------------------
__global__ void __launch_bounds__(256)
gemm_out(const __bf16* __restrict__ wWbf, const __bf16* __restrict__ ctxT,
         float* __restrict__ out) {
    const int wave = threadIdx.x >> 5, lane = threadIdx.x & 31;
    const int unit = blockIdx.x * 8 + wave;          // 2 * 8 * 64 units
    const int n    = unit / (8 * 64);
    const int rem  = unit % (8 * 64);
    const int m0   = (rem % 8) * 64;
    const int p0   = (rem / 8) * 64;
    const int l15  = lane & 15, hi = (lane >= 16);

    v8f acc[4][4] = {};
    for (int k0 = 0; k0 < VC; k0 += 32) {
        v16bf a[4], b[4];
#pragma unroll
        for (int mi = 0; mi < 4; ++mi)
            a[mi] = load_a16(wWbf + (size_t)(m0 + mi * 16 + l15) * VC + k0 + (hi ? 16 : 0));
#pragma unroll
        for (int pi = 0; pi < 4; ++pi)
            b[pi] = load_b16(ctxT + ((size_t)n * HW + p0 + pi * 16 + l15) * VC + k0 + (hi ? 16 : 0));
#pragma unroll
        for (int mi = 0; mi < 4; ++mi)
#pragma unroll
            for (int pi = 0; pi < 4; ++pi)
                acc[mi][pi] = wmma_bf16(a[mi], b[pi], acc[mi][pi]);
    }
#pragma unroll
    for (int mi = 0; mi < 4; ++mi)
#pragma unroll
        for (int pi = 0; pi < 4; ++pi)
#pragma unroll
            for (int r = 0; r < 8; ++r) {
                int m = m0 + mi * 16 + r + (hi ? 8 : 0);
                out[((size_t)n * COUT + m) * HW + p0 + pi * 16 + l15] = acc[mi][pi][r];
            }
}

// ---------------------------------------------------------------------------
// Host launcher
// ---------------------------------------------------------------------------
extern "C" void kernel_launch(void* const* d_in, const int* in_sizes, int n_in,
                              void* d_out, int out_size, void* d_ws, size_t ws_size,
                              hipStream_t stream) {
    (void)in_sizes; (void)n_in; (void)out_size; (void)ws_size;
    const float* x      = (const float*)d_in[0];
    const float* wq     = (const float*)d_in[1];
    const float* gq     = (const float*)d_in[2];
    const float* bq     = (const float*)d_in[3];
    const float* mq     = (const float*)d_in[4];
    const float* vq     = (const float*)d_in[5];
    const float* wk     = (const float*)d_in[6];
    const float* gk     = (const float*)d_in[7];
    const float* bk     = (const float*)d_in[8];
    const float* mk     = (const float*)d_in[9];
    const float* vk     = (const float*)d_in[10];
    const float* wv     = (const float*)d_in[11];
    const float* wpspk  = (const float*)d_in[12];
    const float* bpspk  = (const float*)d_in[13];
    const float* wpspv  = (const float*)d_in[14];
    const float* bpspv  = (const float*)d_in[15];
    const float* wW     = (const float*)d_in[16];
    float* out = (float*)d_out;

    // Workspace carve-up (aligned to 256B)
    size_t off = 0;
    auto carve = [&](size_t bytes) -> void* {
        void* p = (char*)d_ws + off;
        off += (bytes + 255) & ~(size_t)255;
        return p;
    };
    __bf16* W3bf      = (__bf16*)carve((size_t)768 * CIN * 2);
    __bf16* wpspk_bf  = (__bf16*)carve((size_t)256 * 1024 * 2);
    __bf16* wpspv_bf  = (__bf16*)carve((size_t)256 * 1024 * 2);
    __bf16* wWbf      = (__bf16*)carve((size_t)COUT * VC * 2);
    float*  bnq_s     = (float*)carve(256 * 4);
    float*  bnq_b     = (float*)carve(256 * 4);
    float*  bnk_s     = (float*)carve(256 * 4);
    float*  bnk_b     = (float*)carve(256 * 4);
    __bf16* qT        = (__bf16*)carve((size_t)NB * HW * KC * 2);
    float*  kpre      = (float*)carve((size_t)NB * KC * HW * 4);
    float*  vpre      = (float*)carve((size_t)NB * VC * HW * 4);
    float*  pooled_k  = (float*)carve((size_t)NB * 256 * POOLED * 4);
    float*  pooled_v  = (float*)carve((size_t)NB * 256 * POOLED * 4);
    __bf16* catTk     = (__bf16*)carve((size_t)NB * HW * 1024 * 2);
    __bf16* catTv     = (__bf16*)carve((size_t)NB * HW * 1024 * 2);
    __bf16* kT        = (__bf16*)carve((size_t)NB * HW * KC * 2);
    __bf16* vmat      = (__bf16*)carve((size_t)NB * VC * HW * 2);
    __bf16* ctxT      = (__bf16*)carve((size_t)NB * HW * VC * 2);

    // 1) weight conversion + BN folding
    prep_kernel<<<1536, 256, 0, stream>>>(wq, wk, wv, wpspk, wpspv, wW,
                                          gq, bq, mq, vq, gk, bk, mk, vk,
                                          W3bf, wpspk_bf, wpspv_bf, wWbf,
                                          bnq_s, bnq_b, bnk_s, bnk_b);
    // 2) fused q/k/v 1x1 conv (2*12*64 units / 8 waves)
    gemm_qkv<<<192, 256, 0, stream>>>(x, W3bf, bnq_s, bnq_b, bnk_s, bnk_b,
                                      qT, kpre, vpre);
    // 3) adaptive pooling
    pool_kernel<<<(2 * NB * 256 * POOLED + 255) / 256, 256, 0, stream>>>(
        kpre, vpre, pooled_k, pooled_v);
    // 4) upsample + concat (pixel-major bf16)
    upcat_kernel<<<16384, 256, 0, stream>>>(pooled_k, pooled_v, catTk, catTv);
    // 5) PSP 1x1 convs (2*2*4*64 units / 8 waves)
    gemm_psp<<<128, 256, 0, stream>>>(wpspk_bf, wpspv_bf, bpspk, bpspv,
                                      catTk, catTv, kT, vmat);
    // 6) flash attention (512 query tiles, 2 per block)
    attn_kernel<<<256, 128, 0, stream>>>(qT, kT, vmat, ctxT);
    // 7) final 1x1 conv (2*8*64 units / 8 waves)
    gemm_out<<<128, 256, 0, stream>>>(wWbf, ctxT, out);
}
